// Bevformernet_15015205667217
// MI455X (gfx1250) — matrix operand order repaired and checked by reading, not compiled, not run on hardware
//
#include <hip/hip_runtime.h>

// BEVFormer SpatialCrossAttention / MSDeformAttn3D for MI455X (gfx1250, wave32).
// Dense projections: bf16 operands pre-packed once; strip GEMMs with the weight
// matrix staged in LDS per workgroup (strided 16-bit ds reads for B-fragments, conflict-padded),
// A fragments as 16B global_load_b128, fully unrolled v_wmma_f32_16x16x32_bf16.
// Sampling stage: per-query params staged in LDS; coalesced 128B gathers from
// f32 v_proj (4.3MB, L2 resident).

typedef __attribute__((ext_vector_type(16))) __bf16 v16bf;
typedef __attribute__((ext_vector_type(8)))  __bf16 v8bf;
typedef __attribute__((ext_vector_type(8)))  float  v8f;

#define DEV static __device__ __forceinline__

constexpr int NQ = 10000;   // BEV queries
constexpr int C  = 128;     // channels
constexpr int S  = 6;       // cameras
constexpr int M  = 1400;    // Hf*Wf
constexpr int HEADS = 4;
constexpr int PTS   = 8;
constexpr int HF = 28;
constexpr int WF = 50;
constexpr int DH = C / HEADS;  // 32
constexpr int WCAT = 96;       // concat(w_off 64, w_attn 32) columns
constexpr int LPAD = 8;        // LDS row padding (16B) -> min-conflict strided reads

DEV __bf16 f2bf(float f) {
  union { float f; unsigned u; } v; v.f = f;
  unsigned r = (v.u + 0x7FFFu + ((v.u >> 16) & 1u)) >> 16;  // RNE
  unsigned short h = (unsigned short)r;
  return __builtin_bit_cast(__bf16, h);
}

// ---- prep kernels: pack GEMM operands to bf16 (one-time, bandwidth bound) ----
__global__ void pack_add_kernel(const float* __restrict__ a, const float* __restrict__ b,
                                __bf16* __restrict__ dst, int n) {
  int i = blockIdx.x * blockDim.x + threadIdx.x;
  if (i < n) dst[i] = f2bf(a[i] + b[i]);
}
__global__ void pack_kernel(const float* __restrict__ a, __bf16* __restrict__ dst, int n) {
  int i = blockIdx.x * blockDim.x + threadIdx.x;
  if (i < n) dst[i] = f2bf(a[i]);
}
__global__ void pack_wcat_kernel(const float* __restrict__ w_off, const float* __restrict__ w_attn,
                                 __bf16* __restrict__ dst) {
  int i = blockIdx.x * blockDim.x + threadIdx.x;
  if (i >= C * WCAT) return;
  int k = i / WCAT, j = i % WCAT;
  float v = (j < 64) ? w_off[k * 64 + j] : w_attn[k * 32 + (j - 64)];
  dst[i] = f2bf(v);
}

// ---- stage weight matrix (C x LDB bf16) into padded LDS, whole workgroup ----
template <int LDB>
DEV void stage_W(__bf16* smem, const __bf16* __restrict__ Wbf) {
  constexpr int LDSB = LDB + LPAD;
  for (int idx = threadIdx.x; idx < C * (LDB / 8); idx += 128) {
    const int k = idx / (LDB / 8);
    const int j = (idx % (LDB / 8)) * 8;
    *(v8bf*)(smem + k * LDSB + j) = *(const v8bf*)(Wbf + (size_t)k * LDB + j);
  }
  __syncthreads();
}

// ---- WMMA strip: one wave computes a 16-row strip times NT 16-col tiles. ----
// A-fragment (16-bit A 16x32, ISA 7.12.2): lanes 0-15 hold K=0..7 (halves 0-7)
// and K=16..23 (halves 8-15); lanes 16-31 hold K=8..15 / 24..31 -> two 16B loads.
// B-fragment (dense 32x16): lane%16 = N column, halves follow A's K order ->
// strided ds reads from the padded LDS copy of W.
template <int NT, int LDB>
DEV void wmma_strip(const __bf16* __restrict__ Abf, const __bf16* ldsW,
                    int rowBase, v8f* acc) {
  constexpr int LDSB = LDB + LPAD;
  const int lane = threadIdx.x & 31;
  const int kgrp = (lane >> 4) * 8;
  const __bf16* arow = Abf + (size_t)(rowBase + (lane & 15)) * C;
#pragma unroll
  for (int kc = 0; kc < C; kc += 32) {
    const v8bf alo = *(const v8bf*)(arow + kc + kgrp);
    const v8bf ahi = *(const v8bf*)(arow + kc + 16 + kgrp);
    const v16bf a = __builtin_shufflevector(alo, ahi, 0, 1, 2, 3, 4, 5, 6, 7,
                                            8, 9, 10, 11, 12, 13, 14, 15);
    const __bf16* wrow = ldsW + (kc + kgrp) * LDSB + (lane & 15);
#pragma unroll
    for (int ct = 0; ct < NT; ++ct) {
      v16bf b;
#pragma unroll
      for (int i = 0; i < 8; ++i) {
        b[i] = wrow[i * LDSB + ct * 16];
        b[i + 8] = wrow[(16 + i) * LDSB + ct * 16];
      }
      acc[ct] = __builtin_amdgcn_wmma_f32_16x16x32_bf16(false, a, false, b,
                                                        (short)0, acc[ct], false, false);
    }
  }
}

// ---- Stage 1: [off | attn-logits] = (q+qpos) @ [w_off | w_attn] + bias ----
__global__ void __launch_bounds__(128)
qproj_gemm(const __bf16* __restrict__ qbf, const __bf16* __restrict__ wcat,
           const float* __restrict__ b_off, const float* __restrict__ b_attn,
           float* __restrict__ offbuf, float* __restrict__ awbuf) {
  __shared__ __align__(16) __bf16 smem[C * (WCAT + LPAD)];
  stage_W<WCAT>(smem, wcat);                       // all threads reach the barrier
  const int strip = blockIdx.x * 4 + (threadIdx.x >> 5);
  if (strip >= NQ / 16) return;
  const int rowBase = strip * 16;
  v8f acc[6] = {};
  wmma_strip<6, WCAT>(qbf, smem, rowBase, acc);
  const int lane = threadIdx.x & 31;
  const int colOff = lane & 15, rowOff = (lane >> 4) * 8;
#pragma unroll
  for (int ct = 0; ct < 6; ++ct) {
    float* out; int col, ld; float bv;
    if (ct < 4) { out = offbuf; col = ct * 16 + colOff; ld = 64; bv = b_off[col]; }
    else        { out = awbuf;  col = (ct - 4) * 16 + colOff; ld = 32; bv = b_attn[col]; }
#pragma unroll
    for (int r = 0; r < 8; ++r)
      out[(size_t)(rowBase + rowOff + r) * ld + col] = acc[ct][r] + bv;
  }
}

// ---- Stage 2: softmax over the 8 points of each (query, head), in place ----
__global__ void softmax_kernel(float* __restrict__ aw) {
  const int t = blockIdx.x * blockDim.x + threadIdx.x;
  if (t >= NQ * HEADS) return;
  float* p = aw + (size_t)(t >> 2) * 32 + (t & 3) * PTS;
  float v[PTS], mx = -1e30f;
#pragma unroll
  for (int i = 0; i < PTS; ++i) { v[i] = p[i]; mx = fmaxf(mx, v[i]); }
  float sum = 0.f;
#pragma unroll
  for (int i = 0; i < PTS; ++i) { v[i] = __expf(v[i] - mx); sum += v[i]; }
  const float inv = 1.f / sum;
#pragma unroll
  for (int i = 0; i < PTS; ++i) p[i] = v[i] * inv;
}

// ---- Stage 3: v_proj = value @ w_val + b_val  ((S*M) x C), col = h*32+dh, f32 out ----
__global__ void __launch_bounds__(128)
vproj_gemm(const __bf16* __restrict__ valbf, const __bf16* __restrict__ wvalbf,
           const float* __restrict__ b_val, float* __restrict__ vproj) {
  __shared__ __align__(16) __bf16 smem[C * (C + LPAD)];
  stage_W<C>(smem, wvalbf);
  const int strip = blockIdx.x * 4 + (threadIdx.x >> 5);
  if (strip >= S * M / 16) return;
  const int rowBase = strip * 16;
  v8f acc[8] = {};
  wmma_strip<8, C>(valbf, smem, rowBase, acc);
  const int lane = threadIdx.x & 31;
  const int colOff = lane & 15, rowOff = (lane >> 4) * 8;
#pragma unroll
  for (int ct = 0; ct < 8; ++ct) {
    const int col = ct * 16 + colOff;
    const float bv = b_val[col];
#pragma unroll
    for (int r = 0; r < 8; ++r)
      vproj[(size_t)(rowBase + rowOff + r) * C + col] = acc[ct][r] + bv;
  }
}

// ---- Stage 4: bilinear deformable sampling + attention combine + masked camera mean.
// One block per query n: 6 waves = 6 cameras, lane = channel-within-head (dh).
// Per-query params (off, aw, ref) staged in LDS; emits slots directly as bf16.
__global__ void __launch_bounds__(192)
sample_kernel(const float* __restrict__ refpts,            // (S,N,4,2)
              const unsigned char* __restrict__ bev_mask,  // (S,N,4) bool
              const float* __restrict__ offbuf,            // (N,64) = (N,H,Pd,D,2)
              const float* __restrict__ aw,                // (N,32) = (N,H,P)
              const float* __restrict__ vproj,             // (S*M, C)
              __bf16* __restrict__ slotsbf) {              // (N, C) bf16
  const int s = threadIdx.x >> 5;      // camera
  const int lane = threadIdx.x & 31;   // dh channel
  const int n = blockIdx.x;

  __shared__ float red[S][C];
  __shared__ float redm[S];
  __shared__ float s_off[64];
  __shared__ float s_aw[32];
  __shared__ float s_ref[S][8];

  {  // cooperative stage of per-query parameters (broadcast ds reads afterwards)
    const int t = threadIdx.x;
    if (t < 64)        s_off[t] = offbuf[(size_t)n * 64 + t];
    else if (t < 96)   s_aw[t - 64] = aw[(size_t)n * 32 + (t - 64)];
    else if (t < 144) {
      const int ss = (t - 96) >> 3, j = (t - 96) & 7;
      s_ref[ss][j] = refpts[(size_t)(ss * NQ + n) * 8 + j];
    }
  }
  const unsigned char* bm = bev_mask + (size_t)(s * NQ + n) * 4;
  const float maskf = (bm[0] | bm[1] | bm[2] | bm[3]) ? 1.f : 0.f;
  const float* vbase = vproj + (size_t)s * M * C;
  __syncthreads();

  float acc[HEADS] = {0.f, 0.f, 0.f, 0.f};
#pragma unroll
  for (int h = 0; h < HEADS; ++h) {
    const float* vh = vbase + h * DH + lane;
#pragma unroll
    for (int p = 0; p < PTS; ++p) {
      const int d = p & 3, pd = p >> 2;
      const int oi = (((h * 2 + pd) * 4 + d) * 2);
      // ix = (ref.x + off.x/Wf)*Wf - 0.5 = ref.x*Wf + off.x - 0.5
      const float ix = s_ref[s][d * 2 + 0] * (float)WF + s_off[oi + 0] - 0.5f;
      const float iy = s_ref[s][d * 2 + 1] * (float)HF + s_off[oi + 1] - 0.5f;
      const float x0 = floorf(ix), y0 = floorf(iy);
      const float wx1 = ix - x0, wy1 = iy - y0;
      const float wx0 = 1.f - wx1, wy0 = 1.f - wy1;
      float sampv = 0.f;
#pragma unroll
      for (int cy = 0; cy < 2; ++cy) {
#pragma unroll
        for (int cx = 0; cx < 2; ++cx) {
          const float xf = x0 + (float)cx, yf = y0 + (float)cy;
          const bool ok = (xf >= 0.f) && (xf < (float)WF) && (yf >= 0.f) && (yf < (float)HF);
          const int xi = (int)fminf(fmaxf(xf, 0.f), (float)(WF - 1));
          const int yi = (int)fminf(fmaxf(yf, 0.f), (float)(HF - 1));
          const float g = ok ? vh[(size_t)(yi * WF + xi) * C] : 0.f;
          sampv += g * (cx ? wx1 : wx0) * (cy ? wy1 : wy0);
        }
      }
      acc[h] += sampv * s_aw[h * PTS + p];
    }
  }
#pragma unroll
  for (int h = 0; h < HEADS; ++h) red[s][h * DH + lane] = acc[h] * maskf;
  if (lane == 0) redm[s] = maskf;
  __syncthreads();
  if (threadIdx.x < C) {
    const int c = threadIdx.x;
    float sum = 0.f, cnt = 0.f;
#pragma unroll
    for (int s2 = 0; s2 < S; ++s2) { sum += red[s2][c]; cnt += redm[s2]; }
    slotsbf[(size_t)n * C + c] = f2bf(sum / fmaxf(cnt, 1.f));
  }
}

// ---- Stage 5: out = slots @ w_out + b_out + query ----
__global__ void __launch_bounds__(128)
outproj_gemm(const __bf16* __restrict__ slotsbf, const __bf16* __restrict__ woutbf,
             const float* __restrict__ b_out, const float* __restrict__ query,
             float* __restrict__ out) {
  __shared__ __align__(16) __bf16 smem[C * (C + LPAD)];
  stage_W<C>(smem, woutbf);
  const int strip = blockIdx.x * 4 + (threadIdx.x >> 5);
  if (strip >= NQ / 16) return;
  const int rowBase = strip * 16;
  v8f acc[8] = {};
  wmma_strip<8, C>(slotsbf, smem, rowBase, acc);
  const int lane = threadIdx.x & 31;
  const int colOff = lane & 15, rowOff = (lane >> 4) * 8;
#pragma unroll
  for (int ct = 0; ct < 8; ++ct) {
    const int col = ct * 16 + colOff;
    const float bv = b_out[col];
#pragma unroll
    for (int r = 0; r < 8; ++r) {
      const size_t idx = (size_t)(rowBase + rowOff + r) * C + col;
      out[idx] = acc[ct][r] + bv + query[idx];
    }
  }
}

extern "C" void kernel_launch(void* const* d_in, const int* in_sizes, int n_in,
                              void* d_out, int out_size, void* d_ws, size_t ws_size,
                              hipStream_t stream) {
  const float* query  = (const float*)d_in[0];
  // d_in[1] = key : unused by the reference computation
  const float* value  = (const float*)d_in[2];   // (S,M,1,C) contiguous == (S*M, C)
  const float* qpos   = (const float*)d_in[3];
  const float* refpts = (const float*)d_in[4];   // (S,1,N,4,2)
  const unsigned char* bev_mask = (const unsigned char*)d_in[5];  // (S,1,N,4) bool
  const float* w_off  = (const float*)d_in[6];
  const float* b_off  = (const float*)d_in[7];
  const float* w_attn = (const float*)d_in[8];
  const float* b_attn = (const float*)d_in[9];
  const float* w_val  = (const float*)d_in[10];
  const float* b_val  = (const float*)d_in[11];
  const float* w_out  = (const float*)d_in[12];
  const float* b_out  = (const float*)d_in[13];
  (void)in_sizes; (void)n_in; (void)out_size; (void)ws_size;

  // Workspace layout: f32 region, then bf16 region (all 16B aligned).
  float* wsf    = (float*)d_ws;
  float* offbuf = wsf;                          // NQ*64 f32
  float* awbuf  = offbuf + (size_t)NQ * 64;     // NQ*32 f32 (logits -> softmax in place)
  float* vproj  = awbuf  + (size_t)NQ * 32;     // S*M*C f32
  __bf16* bfbase = (__bf16*)(vproj + (size_t)S * M * C);
  __bf16* qbf     = bfbase;                         // NQ*C
  __bf16* valbf   = qbf    + (size_t)NQ * C;        // S*M*C
  __bf16* slotsbf = valbf  + (size_t)S * M * C;     // NQ*C
  __bf16* wcatbf  = slotsbf + (size_t)NQ * C;       // C*96
  __bf16* wvalbf  = wcatbf + (size_t)C * WCAT;      // C*C
  __bf16* woutbf  = wvalbf + (size_t)C * C;         // C*C
  float* outp = (float*)d_out;

  // -- pack operands to bf16 --
  pack_add_kernel<<<(NQ * C + 255) / 256, 256, 0, stream>>>(query, qpos, qbf, NQ * C);
  pack_kernel<<<(S * M * C + 255) / 256, 256, 0, stream>>>(value, valbf, S * M * C);
  pack_wcat_kernel<<<(C * WCAT + 255) / 256, 256, 0, stream>>>(w_off, w_attn, wcatbf);
  pack_kernel<<<(C * C + 255) / 256, 256, 0, stream>>>(w_val, wvalbf, C * C);
  pack_kernel<<<(C * C + 255) / 256, 256, 0, stream>>>(w_out, woutbf, C * C);

  // -- pipeline --
  qproj_gemm<<<(NQ / 16 + 3) / 4, 128, 0, stream>>>(qbf, wcatbf, b_off, b_attn, offbuf, awbuf);
  softmax_kernel<<<(NQ * HEADS + 255) / 256, 256, 0, stream>>>(awbuf);
  vproj_gemm<<<(S * M / 16 + 3) / 4, 128, 0, stream>>>(valbf, wvalbf, b_val, vproj);
  sample_kernel<<<NQ, 192, 0, stream>>>(refpts, bev_mask, offbuf, awbuf, vproj, slotsbf);
  outproj_gemm<<<(NQ / 16 + 3) / 4, 128, 0, stream>>>(slotsbf, woutbf, b_out, query, outp);
}
